// AttentionLayer_8478265442573
// MI455X (gfx1250) — compile-verified, matrix operand order checked by reading, not run on hardware
//
#include <hip/hip_runtime.h>
#include <math.h>

// -------------------------------------------------------------------------
// AttentionLayer for MI455X (gfx1250, wave32, WMMA).
//   B=4, L=4096, D_MODEL=1024, D_HEAD=128.
//   Softmax normalizes over the QUERY axis (per score-matrix column).
// Pipeline (all f16 WMMA with f32 accumulation):
//   K1: QKV projection (f32 -> f16), Vt stored transposed.
//   K2: per-column (over q) online max/sum-exp stats, q-split 8-way.
//   K2b: merge the 8 partial (m,Z) per column.
//   K3: scores -> P = exp(s-m)/Z -> P @ V, k-split 4-way into partial H.
//   K4: reduce the 4 partial H buffers (fixed order -> deterministic).
// Occupancy: K1 24576 waves, K2 8192, K3 4096 (vs ~1024 SIMD32s on chip).
// Workspace: 12.6 MB (QKV f16) + 1 MB (partial stats) + 33.6 MB (partial H).
// -------------------------------------------------------------------------

typedef __attribute__((ext_vector_type(16))) _Float16 v16h;
typedef __attribute__((ext_vector_type(8)))  _Float16 v8h;
typedef __attribute__((ext_vector_type(8)))  float    v8f;
typedef __attribute__((ext_vector_type(4)))  float    v4f;

#define LSEQ   4096
#define DMODEL 1024
#define DHEAD  128
#define NBATCH 4
#define SCALE  0.08838834764831845f   /* 1/sqrt(128) */
#define QSPLIT 8                      /* q-range splits in stats kernel  */
#define KSPLIT 4                      /* k-range splits in output kernel */
#define NCOL   (NBATCH * LSEQ)        /* 16384 score-matrix columns      */
#define HELEMS ((size_t)NBATCH * LSEQ * DHEAD)  /* 2,097,152 */

// ---- WMMA wrapper: D = A(16x32 f16) * B(32x16 f16) + C(16x16 f32) ----
__device__ __forceinline__ v8f wmma_f16(v16h a, v16h b, v8f c) {
  return __builtin_amdgcn_wmma_f32_16x16x32_f16(
      /*neg_a=*/false, a, /*neg_b=*/false, b,
      /*c_mod=*/(short)0, c, /*reuse_a=*/false, /*reuse_b=*/false);
}

// Load a 16x32 f16 tile in A-layout from a row-major f16 matrix.
// ISA layout (16-bit A 16x32): lane<16 -> row=lane, K={0..7,16..23};
//                              lane>=16 -> row=lane-16, K={8..15,24..31}.
// Loading ROWS of a row-major matrix this way also produces the mirrored
// B-operand layout, i.e. B = M^T for a row-major M (standard Q*K^T trick).
__device__ __forceinline__ v16h load_tile_f16(const _Float16* __restrict__ base,
                                              int ld, int row0, int k0) {
  const int lane = threadIdx.x & 31;
  const int r    = lane & 15;
  const int koff = (lane >> 4) ? 8 : 0;
  const _Float16* p = base + (size_t)(row0 + r) * ld + k0 + koff;
  v8h lo = *(const v8h*)(p);          // 16B aligned -> global_load_b128
  v8h hi = *(const v8h*)(p + 16);
  v16h o;
#pragma unroll
  for (int i = 0; i < 8; ++i) { o[i] = lo[i]; o[8 + i] = hi[i]; }
  return o;
}

// Same tile shape, but source is f32 (convert to f16 on the fly).
__device__ __forceinline__ v16h load_tile_f32_cvt(const float* __restrict__ base,
                                                  int ld, int row0, int k0) {
  const int lane = threadIdx.x & 31;
  const int r    = lane & 15;
  const int koff = (lane >> 4) ? 8 : 0;
  const float* p = base + (size_t)(row0 + r) * ld + k0 + koff;
  v4f a0 = *(const v4f*)(p);
  v4f a1 = *(const v4f*)(p + 4);
  v4f b0 = *(const v4f*)(p + 16);
  v4f b1 = *(const v4f*)(p + 20);
  v16h o;
#pragma unroll
  for (int i = 0; i < 4; ++i) {
    o[i]      = (_Float16)a0[i];
    o[4 + i]  = (_Float16)a1[i];
    o[8 + i]  = (_Float16)b0[i];
    o[12 + i] = (_Float16)b1[i];
  }
  return o;
}

// ===========================================================================
// Kernel 1: QKV projection.  y = x @ W^T + b  for the three projections.
//   Qh, Kh : [B*L, 128] row-major f16
//   Vt     : [B, 128, L] f16  (transposed so P@V B-tiles load contiguously)
// One wave per 16x16 output tile; K-loop over DMODEL in steps of 32.
// ===========================================================================
__global__ void qkv_proj_kernel(const float* __restrict__ inp_q,
                                const float* __restrict__ inp_k,
                                const float* __restrict__ inp_v,
                                const float* __restrict__ Wq, const float* __restrict__ bq,
                                const float* __restrict__ Wk, const float* __restrict__ bk,
                                const float* __restrict__ Wv, const float* __restrict__ bv,
                                _Float16* __restrict__ Qh,
                                _Float16* __restrict__ Kh,
                                _Float16* __restrict__ Vt) {
  const int wave = (int)((blockIdx.x * blockDim.x + threadIdx.x) >> 5);
  const int nt = wave & 7;                  // 8 tiles over DHEAD=128
  const int rt = (wave >> 3) & 1023;        // 1024 tiles over B*L=16384
  const int w  = wave >> 13;                // 0=Q 1=K 2=V

  const float* X    = (w == 0) ? inp_q : (w == 1) ? inp_k : inp_v;
  const float* W    = (w == 0) ? Wq    : (w == 1) ? Wk    : Wv;
  const float* bias = (w == 0) ? bq    : (w == 1) ? bk    : bv;

  const int m0 = rt * 16;
  const int n0 = nt * 16;

  v8f c = {};
#pragma unroll 2
  for (int k0 = 0; k0 < DMODEL; k0 += 32) {
    v16h a = load_tile_f32_cvt(X, DMODEL, m0, k0);   // activations tile
    v16h b = load_tile_f32_cvt(W, DMODEL, n0, k0);   // W rows -> B = W^T
    c = wmma_f16(a, b, c);
  }

  const int lane = threadIdx.x & 31;
  const int n    = n0 + (lane & 15);
  const float bn = bias[n];
  const int mh   = (lane >> 4) << 3;   // 0 or 8: which M-half this lane holds

  if (w < 2) {
    _Float16* O = (w == 0) ? Qh : Kh;
#pragma unroll
    for (int r = 0; r < 8; ++r) {
      const int m = m0 + mh + r;
      O[(size_t)m * DHEAD + n] = (_Float16)(c[r] + bn);
    }
  } else {
    // V transposed: Vt[b][n][l]; rows of the C tile are contiguous l's.
    const int b_ = m0 >> 12;            // m0 / 4096
    const int l0 = m0 & (LSEQ - 1);
    v8h st;
#pragma unroll
    for (int r = 0; r < 8; ++r) st[r] = (_Float16)(c[r] + bn);
    _Float16* dst = Vt + ((size_t)b_ * DHEAD + n) * LSEQ + l0 + mh;
    *(v8h*)dst = st;                    // 16B-aligned global_store_b128
  }
}

// ===========================================================================
// Kernel 2: partial per-column softmax stats.
//   s[q,k] = (Q[q].K[k]) / sqrt(d);   per column k: running max + sum-exp.
// Each wave owns 16 columns and ONE of QSPLIT q-chunks (512 q rows), keeping
// an online (max, sum) per column using the C-layout: lane holds column
// n=(lane&15) with 8 M values; the two M-halves merge via __shfl_xor(.,16).
// Partials go to Pm/Pz[qs][b*L+k].
// ===========================================================================
__global__ void col_stats_part_kernel(const _Float16* __restrict__ Qh,
                                      const _Float16* __restrict__ Kh,
                                      float* __restrict__ Pm,
                                      float* __restrict__ Pz) {
  const int wave = (int)((blockIdx.x * blockDim.x + threadIdx.x) >> 5);
  const int qs = wave & (QSPLIT - 1);        // q-chunk
  const int kb = (wave >> 3) & 255;          // 256 column-blocks of 16
  const int b  = wave >> 11;                 // batch
  const int k0 = kb * 16;
  const int qbeg = qs * (LSEQ / QSPLIT);
  const int qend = qbeg + (LSEQ / QSPLIT);

  const _Float16* Qb = Qh + (size_t)b * LSEQ * DHEAD;
  const _Float16* Kb = Kh + (size_t)b * LSEQ * DHEAD;

  // K tiles for these 16 columns are loop-invariant: keep in registers.
  const v16h bk0 = load_tile_f16(Kb, DHEAD, k0, 0);
  const v16h bk1 = load_tile_f16(Kb, DHEAD, k0, 32);
  const v16h bk2 = load_tile_f16(Kb, DHEAD, k0, 64);
  const v16h bk3 = load_tile_f16(Kb, DHEAD, k0, 96);

  float m_run = -3.0e38f;
  float z_run = 0.0f;

  for (int q0 = qbeg; q0 < qend; q0 += 16) {
    v8f c = {};
    c = wmma_f16(load_tile_f16(Qb, DHEAD, q0, 0),  bk0, c);
    c = wmma_f16(load_tile_f16(Qb, DHEAD, q0, 32), bk1, c);
    c = wmma_f16(load_tile_f16(Qb, DHEAD, q0, 64), bk2, c);
    c = wmma_f16(load_tile_f16(Qb, DHEAD, q0, 96), bk3, c);

    // Column max over this tile's 16 rows.
    float t = c[0] * SCALE;
#pragma unroll
    for (int r = 1; r < 8; ++r) t = fmaxf(t, c[r] * SCALE);
    t = fmaxf(t, __shfl_xor(t, 16));               // merge M-halves
    const float m_new = fmaxf(m_run, t);

    float z8 = 0.0f;
#pragma unroll
    for (int r = 0; r < 8; ++r) z8 += __expf(c[r] * SCALE - m_new);
    z8 += __shfl_xor(z8, 16);                      // merge M-halves

    z_run = z_run * __expf(m_run - m_new) + z8;    // online renormalize
    m_run = m_new;
  }

  const int lane = threadIdx.x & 31;
  if (lane < 16) {
    const size_t col = (size_t)b * LSEQ + k0 + lane;
    Pm[(size_t)qs * NCOL + col] = m_run;
    Pz[(size_t)qs * NCOL + col] = z_run;
  }
}

// ===========================================================================
// Kernel 2b: merge QSPLIT partial (m,Z) per column (fixed order).
// ===========================================================================
__global__ void stats_merge_kernel(const float* __restrict__ Pm,
                                   const float* __restrict__ Pz,
                                   float* __restrict__ Mcol,
                                   float* __restrict__ Zcol) {
  const int idx = (int)(blockIdx.x * blockDim.x + threadIdx.x);
  if (idx >= NCOL) return;
  float m = -3.0e38f;
#pragma unroll
  for (int s = 0; s < QSPLIT; ++s) m = fmaxf(m, Pm[(size_t)s * NCOL + idx]);
  float z = 0.0f;
#pragma unroll
  for (int s = 0; s < QSPLIT; ++s)
    z += Pz[(size_t)s * NCOL + idx] * __expf(Pm[(size_t)s * NCOL + idx] - m);
  Mcol[idx] = m;
  Zcol[idx] = z;
}

// ===========================================================================
// Kernel 3: partial H[q,:] = sum_{k in chunk} exp(s[q,k]-m[k])/Z[k] * V[k,:]
// One wave per (16-query tile, k-chunk of 1024), full 128-wide head dim
// resident (8 f32 accumulator tiles). Per 32-k step: 8 score WMMAs ->
// exp/normalize -> stage P f16 through this wave's private LDS region
// (same-wave LDS ops are in-order; regions disjoint across waves, so no
// barrier) -> reload in A-layout via ds_load_b128 -> 8 P@V WMMAs.
// ===========================================================================
__global__ void attn_out_part_kernel(const _Float16* __restrict__ Qh,
                                     const _Float16* __restrict__ Kh,
                                     const _Float16* __restrict__ Vt,
                                     const float* __restrict__ Mcol,
                                     const float* __restrict__ Zcol,
                                     float* __restrict__ Hp) {
  __shared__ __align__(16) _Float16 Pl[8][16 * 32];   // 1KB per wave

  const int wave = (int)((blockIdx.x * blockDim.x + threadIdx.x) >> 5);
  const int wv   = threadIdx.x >> 5;
  const int ks = wave & (KSPLIT - 1);     // k-chunk
  const int qt = (wave >> 2) & 255;       // 256 query tiles
  const int b  = wave >> 10;              // batch
  const int q0 = qt * 16;
  const int kbeg = ks * (LSEQ / KSPLIT);
  const int kend = kbeg + (LSEQ / KSPLIT);

  const _Float16* Qb = Qh + (size_t)b * LSEQ * DHEAD;
  const _Float16* Kb = Kh + (size_t)b * LSEQ * DHEAD;
  const _Float16* Vb = Vt + (size_t)b * DHEAD * LSEQ;
  const float*    Mb = Mcol + (size_t)b * LSEQ;
  const float*    Zb = Zcol + (size_t)b * LSEQ;

  // Q tile (16 x 128) resident for the whole k sweep.
  const v16h aq0 = load_tile_f16(Qb, DHEAD, q0, 0);
  const v16h aq1 = load_tile_f16(Qb, DHEAD, q0, 32);
  const v16h aq2 = load_tile_f16(Qb, DHEAD, q0, 64);
  const v16h aq3 = load_tile_f16(Qb, DHEAD, q0, 96);

  v8f h[8];
#pragma unroll
  for (int i = 0; i < 8; ++i) h[i] = (v8f){};

  const int lane = threadIdx.x & 31;
  const int col  = lane & 15;
  const int mh   = (lane >> 4) << 3;
  _Float16* myP  = &Pl[wv][0];

  for (int k0 = kbeg; k0 < kend; k0 += 32) {
    // Pull the next K chunk toward the caches while we compute this one.
    if (k0 + 32 < kend)
      __builtin_prefetch(Kb + (size_t)(k0 + 32) * DHEAD, 0, 1);

#pragma unroll
    for (int sub = 0; sub < 2; ++sub) {
      const int kc = k0 + sub * 16;
      v8f c = {};
      c = wmma_f16(aq0, load_tile_f16(Kb, DHEAD, kc, 0),  c);
      c = wmma_f16(aq1, load_tile_f16(Kb, DHEAD, kc, 32), c);
      c = wmma_f16(aq2, load_tile_f16(Kb, DHEAD, kc, 64), c);
      c = wmma_f16(aq3, load_tile_f16(Kb, DHEAD, kc, 96), c);

      const float mc = Mb[kc + col];
      const float zi = 1.0f / Zb[kc + col];
#pragma unroll
      for (int r = 0; r < 8; ++r) {
        const float p = __expf(c[r] * SCALE - mc) * zi;
        myP[(mh + r) * 32 + sub * 16 + col] = (_Float16)p;   // C -> row-major
      }
    }

    // Reload P (16x32) in A-layout from LDS (same-wave LDS ops are in-order).
    const v16h pa = load_tile_f16(myP, 32, 0, 0);

#pragma unroll
    for (int vt = 0; vt < 8; ++vt) {
      // B[k][v] = V[k0+k, 16*vt+n] = Vt[16*vt+n][k0+k]: contiguous row loads.
      const v16h bvt = load_tile_f16(Vb, LSEQ, vt * 16, k0);
      h[vt] = wmma_f16(pa, bvt, h[vt]);
    }
  }

  // Store partial H (f32, row-major [B, L, DHEAD]) for this k-chunk.
  float* O = Hp + (size_t)ks * HELEMS + ((size_t)b * LSEQ + q0) * DHEAD;
#pragma unroll
  for (int vt = 0; vt < 8; ++vt) {
#pragma unroll
    for (int r = 0; r < 8; ++r) {
      O[(size_t)(mh + r) * DHEAD + vt * 16 + col] = h[vt][r];
    }
  }
}

// ===========================================================================
// Kernel 4: out = sum over KSPLIT partial-H buffers (fixed summation order,
// vectorized b128 loads/stores).
// ===========================================================================
__global__ void reduce_h_kernel(const float* __restrict__ Hp,
                                float* __restrict__ out) {
  const size_t i4 = ((size_t)blockIdx.x * blockDim.x + threadIdx.x) * 4;
  if (i4 >= HELEMS) return;
  v4f acc = *(const v4f*)(Hp + i4);
#pragma unroll
  for (int s = 1; s < KSPLIT; ++s) {
    v4f t = *(const v4f*)(Hp + (size_t)s * HELEMS + i4);
    acc += t;
  }
  *(v4f*)(out + i4) = acc;
}

// ===========================================================================
extern "C" void kernel_launch(void* const* d_in, const int* in_sizes, int n_in,
                              void* d_out, int out_size, void* d_ws, size_t ws_size,
                              hipStream_t stream) {
  const float* inp_q = (const float*)d_in[0];
  const float* inp_k = (const float*)d_in[1];
  const float* inp_v = (const float*)d_in[2];
  const float* Wq    = (const float*)d_in[3];
  const float* bq    = (const float*)d_in[4];
  const float* Wk    = (const float*)d_in[5];
  const float* bk    = (const float*)d_in[6];
  const float* Wv    = (const float*)d_in[7];
  const float* bv    = (const float*)d_in[8];
  float* out = (float*)d_out;

  // Workspace partition (~47 MB total).
  char* ws = (char*)d_ws;
  _Float16* Qh = (_Float16*)ws;  ws += HELEMS * sizeof(_Float16);
  _Float16* Kh = (_Float16*)ws;  ws += HELEMS * sizeof(_Float16);
  _Float16* Vt = (_Float16*)ws;  ws += HELEMS * sizeof(_Float16);
  float* Mcol  = (float*)ws;     ws += (size_t)NCOL * sizeof(float);
  float* Zcol  = (float*)ws;     ws += (size_t)NCOL * sizeof(float);
  float* Pm    = (float*)ws;     ws += (size_t)QSPLIT * NCOL * sizeof(float);
  float* Pz    = (float*)ws;     ws += (size_t)QSPLIT * NCOL * sizeof(float);
  float* Hp    = (float*)ws;     // KSPLIT * HELEMS floats

  // K1: 3 mats * 1024 row-tiles * 8 col-tiles = 24576 waves -> 3072 blocks.
  qkv_proj_kernel<<<3072, 256, 0, stream>>>(inp_q, inp_k, inp_v,
                                            Wq, bq, Wk, bk, Wv, bv,
                                            Qh, Kh, Vt);
  // K2: 4 * 256 * QSPLIT = 8192 waves -> 1024 blocks.
  col_stats_part_kernel<<<1024, 256, 0, stream>>>(Qh, Kh, Pm, Pz);
  // K2b: one thread per column.
  stats_merge_kernel<<<NCOL / 256, 256, 0, stream>>>(Pm, Pz, Mcol, Zcol);
  // K3: 4 * 256 * KSPLIT = 4096 waves -> 512 blocks.
  attn_out_part_kernel<<<512, 256, 0, stream>>>(Qh, Kh, Vt, Mcol, Zcol, Hp);
  // K4: vectorized reduction, HELEMS/4 threads.
  reduce_h_kernel<<<(int)(HELEMS / 4 / 256), 256, 0, stream>>>(Hp, out);
}
